// MultiHeadAttention_66400194396667
// MI455X (gfx1250) — matrix operand order repaired and checked by reading, not compiled, not run on hardware
//
#include <hip/hip_runtime.h>
#include <hip/hip_bf16.h>

// ---------------------------------------------------------------------------
// MI455X (gfx1250) multi-head attention, bf16 WMMA path + async LDS staging.
//   B=4, S=2048, D=1024, H=16, HD=64
//   d_in: xq, xk, xv, mask(ignored: causal), W_q, W_k, W_v, W_o   (all f32)
//   d_out: [B,S,D] f32
// ---------------------------------------------------------------------------

#define Bn   4
#define Sn   2048
#define Dn   1024
#define Hn   16
#define HDn  64
#define Mn   (Bn * Sn)          // 8192 rows in projection GEMMs

typedef __attribute__((ext_vector_type(16))) __bf16 v16bf;
typedef __attribute__((ext_vector_type(8)))  __bf16 v8bf;
typedef __attribute__((ext_vector_type(8)))  float  v8f;
typedef int v4i_gcc __attribute__((vector_size(16)));   // matches builtin sig

#if defined(__has_builtin)
#if __has_builtin(__builtin_amdgcn_global_load_async_to_lds_b128)
#define HAVE_ASYNC_LDS 1
#endif
#endif
#ifndef HAVE_ASYNC_LDS
#define HAVE_ASYNC_LDS 0
#endif

__device__ __forceinline__ v8f vzero8() {
  v8f z;
#pragma unroll
  for (int i = 0; i < 8; ++i) z[i] = 0.0f;
  return z;
}

__device__ __forceinline__ v8f wmma_bf16(v16bf a, v16bf b, v8f c) {
  return __builtin_amdgcn_wmma_f32_16x16x32_bf16(
      /*neg_a=*/false, a, /*neg_b=*/false, b,
      /*c_mod=*/(short)0, c, /*reuse_a=*/false, /*reuse_b=*/false);
}

// Load a 16-element bf16 fragment as two 16B chunks (-> two *_load_b128).
__device__ __forceinline__ v16bf ld_frag2(const __bf16* p0, const __bf16* p1) {
  v8bf a = *(const v8bf*)p0;
  v8bf b = *(const v8bf*)p1;
  return __builtin_shufflevector(a, b, 0, 1, 2, 3, 4, 5, 6, 7,
                                 8, 9, 10, 11, 12, 13, 14, 15);
}

// 16B global -> LDS copy. Async DMA path on gfx1250 (no VGPR round trip),
// synchronous vector copy otherwise.
__device__ __forceinline__ void async_copy16(void* lds, const void* g) {
#if HAVE_ASYNC_LDS
  __builtin_amdgcn_global_load_async_to_lds_b128(
      (__attribute__((address_space(1))) v4i_gcc*)(void*)g,
      (__attribute__((address_space(3))) v4i_gcc*)lds,
      /*offset=*/0, /*cpol=*/0);
#else
  *(v8bf*)lds = *(const v8bf*)g;
#endif
}

__device__ __forceinline__ void wait_async_all() {
#if HAVE_ASYNC_LDS
#if __has_builtin(__builtin_amdgcn_s_wait_asynccnt)
  __builtin_amdgcn_s_wait_asynccnt(0);
#else
  asm volatile("s_wait_asynccnt 0" ::: "memory");
#endif
#endif
}

// Wave-local LDS producer->consumer sync (cross-lane, same wave).
__device__ __forceinline__ void wave_lds_sync() {
  asm volatile("s_wait_dscnt 0" ::: "memory");
  __builtin_amdgcn_wave_barrier();
}

// ---------------------------------------------------------------------------
// Projection GEMM:  Y[m,n] = sum_k X[m,k] * W[k,n]  (f32 in, bf16 out)
// Output head-major [B,H,S,HD]; Q gets (1/sqrt(HD))*log2(e) folded in.
// Block tile 128x64, 8 waves, 32x32 per wave, K-step 32.
// ---------------------------------------------------------------------------
__global__ __launch_bounds__(256) void proj_qkv_kernel(
    const float* __restrict__ X,   // [Mn, Dn]
    const float* __restrict__ W,   // [Dn, H*HD]
    __bf16* __restrict__ Out,      // [B,H,S,HD]
    float scale) {
  __shared__ __align__(16) __bf16 As[128][40];   // rows 80B (16B aligned)
  __shared__ __align__(16) __bf16 BsT[64][40];   // [n][k], rows 80B

  const int tid  = threadIdx.x;
  const int lane = tid & 31;
  const int w    = tid >> 5;
  const int wm   = w & 3;
  const int wn   = w >> 2;
  const int hi   = lane >> 4;
  const int ln   = lane & 15;
  const int m0   = blockIdx.x * 128;
  const int n0   = blockIdx.y * 64;

  v8f acc[2][2];
#pragma unroll
  for (int mi = 0; mi < 2; ++mi)
#pragma unroll
    for (int ni = 0; ni < 2; ++ni) acc[mi][ni] = vzero8();

  for (int k0 = 0; k0 < Dn; k0 += 32) {
    // stage A 128x32 (f32 -> bf16), float4 global loads, b64 LDS stores
#pragma unroll
    for (int i = 0; i < 4; ++i) {
      int idx = tid + i * 256;
      int r = idx >> 3, c4 = idx & 7;
      float4 v = *(const float4*)(X + (size_t)(m0 + r) * Dn + k0 + c4 * 4);
      __bf16* dst = &As[r][c4 * 4];
      dst[0] = (__bf16)v.x; dst[1] = (__bf16)v.y;
      dst[2] = (__bf16)v.z; dst[3] = (__bf16)v.w;
    }
    // stage B 32x64 transposed into BsT[n][k]
#pragma unroll
    for (int i = 0; i < 2; ++i) {
      int idx = tid + i * 256;
      int r = idx >> 4, c4 = idx & 15;
      float4 v = *(const float4*)(W + (size_t)(k0 + r) * (Hn * HDn) + n0 + c4 * 4);
      BsT[c4 * 4 + 0][r] = (__bf16)v.x;
      BsT[c4 * 4 + 1][r] = (__bf16)v.y;
      BsT[c4 * 4 + 2][r] = (__bf16)v.z;
      BsT[c4 * 4 + 3][r] = (__bf16)v.w;
    }
    __syncthreads();

    v16bf af[2], bfr[2];
#pragma unroll
    for (int mi = 0; mi < 2; ++mi) {
      const __bf16* ap = &As[wm * 32 + mi * 16 + ln][0];
      af[mi] = ld_frag2(ap + (hi << 3), ap + 16 + (hi << 3));
    }
#pragma unroll
    for (int ni = 0; ni < 2; ++ni) {
      const __bf16* bp = &BsT[wn * 32 + ni * 16 + ln][hi << 3];
      bfr[ni] = ld_frag2(bp, bp + 16);
    }
#pragma unroll
    for (int mi = 0; mi < 2; ++mi)
#pragma unroll
      for (int ni = 0; ni < 2; ++ni)
        acc[mi][ni] = wmma_bf16(af[mi], bfr[ni], acc[mi][ni]);
    __syncthreads();
  }

#pragma unroll
  for (int mi = 0; mi < 2; ++mi)
#pragma unroll
    for (int ni = 0; ni < 2; ++ni)
#pragma unroll
      for (int r = 0; r < 8; ++r) {
        int m = m0 + wm * 32 + mi * 16 + r + (hi << 3);
        int n = n0 + wn * 32 + ni * 16 + ln;
        int b = m >> 11, s = m & (Sn - 1);
        int h = n >> 6, hd = n & (HDn - 1);
        Out[(((size_t)(b * Hn + h) * Sn + s) << 6) + hd] =
            (__bf16)(acc[mi][ni][r] * scale);
      }
}

// ---------------------------------------------------------------------------
// Flash attention. Grid: (S/128, B*H). 8 waves/block, 16 query rows/wave.
// K tile staged with async global->LDS DMA; V tile transposed via VGPRs.
// ---------------------------------------------------------------------------
__global__ __launch_bounds__(256) void attn_kernel(
    const __bf16* __restrict__ Q,   // [B*H, S, HD]
    const __bf16* __restrict__ Kk,
    const __bf16* __restrict__ V,
    __bf16* __restrict__ Ctx) {
  __shared__ __align__(16) __bf16 Kt[32][72];       // [key][dim], rows 144B
  __shared__ __align__(16) __bf16 VtT[64][40];      // [dim][key], rows 80B
  __shared__ __align__(16) float  sc_lds[8][16][36];// raw scores, rows 144B
  __shared__ __align__(16) __bf16 P_lds[8][16][40]; // exp'd probs, rows 80B
  __shared__ float alpha_lds[8][16];

  const int tid  = threadIdx.x;
  const int lane = tid & 31;
  const int w    = tid >> 5;
  const int hi   = lane >> 4;
  const int ln   = lane & 15;
  const int q0   = blockIdx.x * 128;
  const int bh   = blockIdx.y;
  const int qw   = q0 + w * 16;

  const __bf16* Qb = Q  + (size_t)bh * Sn * HDn;
  const __bf16* Kb = Kk + (size_t)bh * Sn * HDn;
  const __bf16* Vb = V  + (size_t)bh * Sn * HDn;

  // Q fragments: 16x64 = two 16x32 A-fragments (4x global_load_b128)
  v16bf qf[2];
  {
    const __bf16* qp = Qb + (size_t)(qw + ln) * HDn;
#pragma unroll
    for (int c = 0; c < 2; ++c)
      qf[c] = ld_frag2(qp + c * 32 + (hi << 3), qp + c * 32 + 16 + (hi << 3));
  }

  v8f ctx[4];
#pragma unroll
  for (int ni = 0; ni < 4; ++ni) ctx[ni] = vzero8();
  float m_r = -1e30f, l_r = 0.0f;   // lane (hi,ln) tracks row ln (replicated)

  const int kend = (q0 + 128 < Sn) ? (q0 + 128) : Sn;
  for (int k0 = 0; k0 < kend; k0 += 32) {
    // ---- K tile: one async 16B DMA per thread (32x64 bf16 = 4KB) ----
    {
      int kk = tid >> 3, c = tid & 7;
      async_copy16(&Kt[kk][c * 8], Kb + (size_t)(k0 + kk) * HDn + c * 8);
    }
    // ---- V tile: transpose through VGPRs (overlaps the async DMA) ----
    {
      const unsigned int* vg = (const unsigned int*)(Vb + (size_t)k0 * HDn);
#pragma unroll
      for (int i = 0; i < 4; ++i) {
        int e = tid + i * 256;            // dword index over 32x64 bf16
        int kv = e >> 5, cd = e & 31;
        union { unsigned int u; __bf16 h[2]; } cv;
        cv.u = vg[e];
        VtT[2 * cd + 0][kv] = cv.h[0];
        VtT[2 * cd + 1][kv] = cv.h[1];
      }
    }
    wait_async_all();
    __syncthreads();

    if (k0 <= qw + 15) {                 // wave-uniform causal skip
      // ---- scores: Q(16x64) . K_tile^T ----
      v8f s0 = vzero8(), s1 = vzero8();
#pragma unroll
      for (int c = 0; c < 2; ++c) {
        const __bf16* kp = &Kt[ln][(hi << 3) + c * 32];
        s0 = wmma_bf16(qf[c], ld_frag2(kp, kp + 16), s0);
      }
#pragma unroll
      for (int c = 0; c < 2; ++c) {
        const __bf16* kp = &Kt[ln + 16][(hi << 3) + c * 32];
        s1 = wmma_bf16(qf[c], ld_frag2(kp, kp + 16), s1);
      }

      // ---- causal mask only on diagonal-straddling blocks ----
      if (k0 + 31 > qw) {
#pragma unroll
        for (int r = 0; r < 8; ++r) {
          int qrow = qw + r + (hi << 3);
          int key0 = k0 + ln;
          if (key0 > qrow)      s0[r] = -1e30f;
          if (key0 + 16 > qrow) s1[r] = -1e30f;
        }
      }
      // spill raw scores (C layout -> row-major)
#pragma unroll
      for (int r = 0; r < 8; ++r) {
        sc_lds[w][r + (hi << 3)][ln]      = s0[r];
        sc_lds[w][r + (hi << 3)][16 + ln] = s1[r];
      }
      wave_lds_sync();

      // ---- distributed online softmax: lane (hi,ln) does row ln, half hi ----
      {
        const float4* s4 = (const float4*)&sc_lds[w][ln][hi << 4];
        float sv[16];
#pragma unroll
        for (int j = 0; j < 4; ++j) {
          float4 t = s4[j];
          sv[4 * j + 0] = t.x; sv[4 * j + 1] = t.y;
          sv[4 * j + 2] = t.z; sv[4 * j + 3] = t.w;
        }
        float mloc = sv[0];
#pragma unroll
        for (int j = 1; j < 16; ++j) mloc = fmaxf(mloc, sv[j]);
        mloc = fmaxf(mloc, __shfl_xor(mloc, 16));
        float mnew  = fmaxf(m_r, mloc);
        float alpha = exp2f(m_r - mnew);
        float sum = 0.0f;
        v8bf p0, p1;
#pragma unroll
        for (int j = 0; j < 8; ++j) {
          float p = exp2f(sv[j] - mnew);
          sum += p;
          p0[j] = (__bf16)p;
        }
#pragma unroll
        for (int j = 0; j < 8; ++j) {
          float p = exp2f(sv[8 + j] - mnew);
          sum += p;
          p1[j] = (__bf16)p;
        }
        sum += __shfl_xor(sum, 16);
        l_r = alpha * l_r + sum;
        m_r = mnew;
        v8bf* pp = (v8bf*)&P_lds[w][ln][hi << 4];
        pp[0] = p0;
        pp[1] = p1;
        if (hi == 0) alpha_lds[w][ln] = alpha;
      }
      wave_lds_sync();

      // ---- rescale running context ----
      float al[8];
#pragma unroll
      for (int r = 0; r < 8; ++r) al[r] = alpha_lds[w][r + (hi << 3)];
#pragma unroll
      for (int ni = 0; ni < 4; ++ni)
#pragma unroll
        for (int r = 0; r < 8; ++r) ctx[ni][r] *= al[r];

      // ---- ctx += P(16x32) . V_tile(32x64) ----
      const __bf16* prow = &P_lds[w][ln][0];
      v16bf pf = ld_frag2(prow + (hi << 3), prow + 16 + (hi << 3));
#pragma unroll
      for (int ni = 0; ni < 4; ++ni) {
        const __bf16* vp = &VtT[ni * 16 + ln][hi << 3];
        ctx[ni] = wmma_bf16(pf, ld_frag2(vp, vp + 16), ctx[ni]);
      }
    }
    __syncthreads();
  }

  // ---- normalize by 1/l and store ctx (bf16, head-major) ----
  if (lane < 16) alpha_lds[w][lane] = 1.0f / l_r;
  wave_lds_sync();
  float il[8];
#pragma unroll
  for (int r = 0; r < 8; ++r) il[r] = alpha_lds[w][r + (hi << 3)];
  __bf16* cp = Ctx + ((size_t)bh * Sn + qw) * HDn;
#pragma unroll
  for (int ni = 0; ni < 4; ++ni)
#pragma unroll
    for (int r = 0; r < 8; ++r)
      cp[(size_t)(r + (hi << 3)) * HDn + ni * 16 + ln] =
          (__bf16)(ctx[ni][r] * il[r]);
}

// ---------------------------------------------------------------------------
// Output projection: out[m,n] = sum_f ctx[m,f] * W_o[f,n]
// A tile (already bf16) staged via async global->LDS DMA.
// ---------------------------------------------------------------------------
__global__ __launch_bounds__(256) void out_proj_kernel(
    const __bf16* __restrict__ Ctx,  // [B,H,S,HD]
    const float* __restrict__ Wo,    // [H*HD, Dn]
    float* __restrict__ Out) {       // [Mn, Dn]
  __shared__ __align__(16) __bf16 As[128][40];
  __shared__ __align__(16) __bf16 BsT[64][40];

  const int tid  = threadIdx.x;
  const int lane = tid & 31;
  const int w    = tid >> 5;
  const int wm   = w & 3;
  const int wn   = w >> 2;
  const int hi   = lane >> 4;
  const int ln   = lane & 15;
  const int m0   = blockIdx.x * 128;
  const int n0   = blockIdx.y * 64;

  v8f acc[2][2];
#pragma unroll
  for (int mi = 0; mi < 2; ++mi)
#pragma unroll
    for (int ni = 0; ni < 2; ++ni) acc[mi][ni] = vzero8();

  for (int k0 = 0; k0 < Hn * HDn; k0 += 32) {
    // stage A 128x32 bf16 from head-major ctx: 2 async 16B DMAs per thread
#pragma unroll
    for (int i = 0; i < 2; ++i) {
      int e = tid + i * 256;
      int r = e >> 2, c = e & 3;
      int m = m0 + r;
      int b = m >> 11, s = m & (Sn - 1);
      const __bf16* src = Ctx +
          (((size_t)(b * Hn + (k0 >> 6)) * Sn + s) << 6) + (k0 & (HDn - 1)) +
          c * 8;
      async_copy16(&As[r][c * 8], src);
    }
    // stage B 32x64 transposed (f32 -> bf16 through VGPRs)
#pragma unroll
    for (int i = 0; i < 2; ++i) {
      int idx = tid + i * 256;
      int r = idx >> 4, c4 = idx & 15;
      float4 v = *(const float4*)(Wo + (size_t)(k0 + r) * Dn + n0 + c4 * 4);
      BsT[c4 * 4 + 0][r] = (__bf16)v.x;
      BsT[c4 * 4 + 1][r] = (__bf16)v.y;
      BsT[c4 * 4 + 2][r] = (__bf16)v.z;
      BsT[c4 * 4 + 3][r] = (__bf16)v.w;
    }
    wait_async_all();
    __syncthreads();

    v16bf af[2], bfr[2];
#pragma unroll
    for (int mi = 0; mi < 2; ++mi) {
      const __bf16* ap = &As[wm * 32 + mi * 16 + ln][0];
      af[mi] = ld_frag2(ap + (hi << 3), ap + 16 + (hi << 3));
    }
#pragma unroll
    for (int ni = 0; ni < 2; ++ni) {
      const __bf16* bp = &BsT[wn * 32 + ni * 16 + ln][hi << 3];
      bfr[ni] = ld_frag2(bp, bp + 16);
    }
#pragma unroll
    for (int mi = 0; mi < 2; ++mi)
#pragma unroll
      for (int ni = 0; ni < 2; ++ni)
        acc[mi][ni] = wmma_bf16(af[mi], bfr[ni], acc[mi][ni]);
    __syncthreads();
  }

#pragma unroll
  for (int mi = 0; mi < 2; ++mi)
#pragma unroll
    for (int ni = 0; ni < 2; ++ni)
#pragma unroll
      for (int r = 0; r < 8; ++r) {
        int m = m0 + wm * 32 + mi * 16 + r + (hi << 3);
        int n = n0 + wn * 32 + ni * 16 + ln;
        Out[(size_t)m * Dn + n] = acc[mi][ni][r];
      }
}

// ---------------------------------------------------------------------------
extern "C" void kernel_launch(void* const* d_in, const int* in_sizes, int n_in,
                              void* d_out, int out_size, void* d_ws,
                              size_t ws_size, hipStream_t stream) {
  (void)in_sizes; (void)n_in; (void)out_size; (void)ws_size;
  const float* xq = (const float*)d_in[0];
  const float* xk = (const float*)d_in[1];
  const float* xv = (const float*)d_in[2];
  // d_in[3] = mask (causal, recomputed on device)
  const float* Wq = (const float*)d_in[4];
  const float* Wk = (const float*)d_in[5];
  const float* Wv = (const float*)d_in[6];
  const float* Wo = (const float*)d_in[7];
  float* out = (float*)d_out;

  const size_t headsz = (size_t)Bn * Hn * Sn * HDn * sizeof(__bf16);  // 16 MB
  char* ws = (char*)d_ws;
  __bf16* qb = (__bf16*)(ws);
  __bf16* kb = (__bf16*)(ws + headsz);
  __bf16* vb = (__bf16*)(ws + 2 * headsz);
  __bf16* cb = (__bf16*)(ws + 3 * headsz);

  const dim3 blk(256);
  const dim3 gproj(Mn / 128, (Hn * HDn) / 64);   // 64 x 16
  const dim3 gattn(Sn / 128, Bn * Hn);           // 16 x 64

  const float qscale = 0.125f * 1.44269504088896340736f;  // 1/sqrt(64)*log2(e)

  hipLaunchKernelGGL(proj_qkv_kernel, gproj, blk, 0, stream, xq, Wq, qb, qscale);
  hipLaunchKernelGGL(proj_qkv_kernel, gproj, blk, 0, stream, xk, Wk, kb, 1.0f);
  hipLaunchKernelGGL(proj_qkv_kernel, gproj, blk, 0, stream, xv, Wv, vb, 1.0f);
  hipLaunchKernelGGL(attn_kernel, gattn, blk, 0, stream, qb, kb, vb, cb);
  hipLaunchKernelGGL(out_proj_kernel, gproj, blk, 0, stream, cb, Wo, out);
}
